// MultiHeadSelfAttention_47588237639930
// MI455X (gfx1250) — compile-verified, matrix operand order checked by reading, not run on hardware
//
#include <hip/hip_runtime.h>

// ---------------- constants (reference: DIM=2048, HEADS=16, B=2, T=2048) ----
constexpr int kDIM   = 2048;
constexpr int kNQKV  = 3 * kDIM;   // 6144
constexpr int kB     = 2;
constexpr int kT     = 2048;
constexpr int kBT    = kB * kT;    // 4096 rows
constexpr int kHEADS = 16;
constexpr int kHD    = kDIM / kHEADS; // 128

typedef __attribute__((ext_vector_type(16))) __bf16 v16bf;
typedef __attribute__((ext_vector_type(8)))  float  v8f;

// gfx1250 async Global->LDS path (ASYNCcnt). Guarded: falls back to direct
// global loads if the builtin is absent (also keeps the host parse trivial).
// Probe-learned signature: (int4 AS1*, int4 AS3*, imm int, imm int).
#if defined(__HIP_DEVICE_COMPILE__) && __has_builtin(__builtin_amdgcn_global_load_async_to_lds_b128)
#define HAVE_ASYNC_LDS 1
typedef int v4i_vs __attribute__((vector_size(16)));
typedef __attribute__((address_space(1))) v4i_vs gv4i;
typedef __attribute__((address_space(3))) v4i_vs lv4i;
#else
#define HAVE_ASYNC_LDS 0
#endif

// ---------------------------------------------------------------------------
// Fragment loaders for V_WMMA_F32_16X16X32_BF16 (wave32).
// A (16x32, bf16): lanes 0-15 M=0..15, lanes 16-31 M=0..15;
//   elem e <-> K = (e&7) + (e>=8)*16 + half*8   (half = lane>>4)
// B (32x16, bf16): lane -> N = lane&15; same K mapping per elem.
// C/D (16x16, f32): VGPR r -> M = r + 8*half; lane -> N = lane&15.
// ---------------------------------------------------------------------------

__device__ __forceinline__ v16bf load_a_frag_f32(const float* __restrict__ src,
                                                 int row0, int ld, int kbase, int lane) {
  int m = lane & 15, half = lane >> 4;
  const float* p = src + (size_t)(row0 + m) * ld + kbase + half * 8;
  v16bf a;
#pragma unroll
  for (int e = 0; e < 8; ++e) {
    a[e]     = (__bf16)p[e];        // K = kbase + half*8 + e
    a[e + 8] = (__bf16)p[e + 16];   // K = kbase + half*8 + 16 + e
  }
  return a;
}

__device__ __forceinline__ v16bf load_a_frag_bf16(const __bf16* __restrict__ src,
                                                  int row0, int ld, int kbase, int lane) {
  int m = lane & 15, half = lane >> 4;
  const __bf16* p = src + (size_t)(row0 + m) * ld + kbase + half * 8;
  v16bf a;
#pragma unroll
  for (int e = 0; e < 8; ++e) {
    a[e]     = p[e];
    a[e + 8] = p[e + 16];
  }
  return a;
}

// B-fragment from f32 row-major src [K x N]. For fixed elem e the 16 lanes of a
// half read 64 consecutive bytes -> two fully-used 64B segments per vmem op.
// Also used on LDS-resident tiles (compiler emits ds_load after inlining).
__device__ __forceinline__ v16bf load_b_frag_f32(const float* __restrict__ src,
                                                 int kbase, int ld, int col0, int lane) {
  int n = lane & 15, half = lane >> 4;
  const float* p = src + (size_t)(kbase + half * 8) * ld + col0 + n;
  v16bf b;
#pragma unroll
  for (int e = 0; e < 8; ++e) {
    b[e]     = (__bf16)p[(size_t)e * ld];
    b[e + 8] = (__bf16)p[(size_t)(e + 16) * ld];
  }
  return b;
}

__device__ __forceinline__ v8f wmma_bf16(v16bf a, v16bf b, v8f c) {
  return __builtin_amdgcn_wmma_f32_16x16x32_bf16(false, a, false, b, (short)0, c,
                                                 false, false);
}

// ---------------------------------------------------------------------------
// Kernel 1: QKV = x @ W_qkv + b_qkv ; scatter Q(bf16->ws), K/V(f32->d_out[B,H,T,D])
// One wave computes a 32x64 tile: 2 A-frags x 4 B-frags -> 8 WMMAs per k-step.
// Grid (6144/64/4 = 24, 4096/32 = 128), block 128 (4 waves).
// ---------------------------------------------------------------------------
__global__ __launch_bounds__(128) void qkv_kernel(
    const float* __restrict__ x, const float* __restrict__ Wqkv,
    const float* __restrict__ bqkv, __bf16* __restrict__ Qbf,
    float* __restrict__ Kout, float* __restrict__ Vout) {
  int lane = threadIdx.x & 31, w = threadIdx.x >> 5;
  int col0 = (blockIdx.x * 4 + w) * 64;   // 0..6080
  int row0 = blockIdx.y * 32;             // 0..4064
  int n = lane & 15, half = lane >> 4;

  v8f acc[2][4] = {};
  for (int kb = 0; kb < kDIM; kb += 32) {
    v16bf a0 = load_a_frag_f32(x, row0,      kDIM, kb, lane);
    v16bf a1 = load_a_frag_f32(x, row0 + 16, kDIM, kb, lane);
    // prefetch next k-slice of the A panel and B panel (global_prefetch_b8)
    __builtin_prefetch(x + (size_t)(row0 + n) * kDIM + kb + 32, 0, 1);
    __builtin_prefetch(Wqkv + (size_t)(kb + 32 + half * 8) * kNQKV + col0 + n * 4, 0, 1);
#pragma unroll
    for (int j = 0; j < 4; ++j) {
      v16bf b = load_b_frag_f32(Wqkv, kb, kNQKV, col0 + j * 16, lane);
      acc[0][j] = wmma_bf16(a0, b, acc[0][j]);
      acc[1][j] = wmma_bf16(a1, b, acc[1][j]);
    }
  }

#pragma unroll
  for (int j = 0; j < 4; ++j) {
    int col = col0 + j * 16 + n;
    float bias = bqkv[col];
    int sec = col >> 11;                 // 0=Q, 1=K, 2=V (uniform per tile)
    int c2  = col & (kDIM - 1);
    int h = c2 >> 7, d = c2 & (kHD - 1);
#pragma unroll
    for (int rb = 0; rb < 2; ++rb) {
#pragma unroll
      for (int r = 0; r < 8; ++r) {
        int row = row0 + rb * 16 + r + half * 8;   // [0,4096)
        float v = acc[rb][j][r] + bias;
        int bi = row >> 11, t = row & (kT - 1);
        size_t idx = (((size_t)bi * kHEADS + h) * kT + t) * kHD + d;
        if (sec == 0)      Qbf[idx]  = (__bf16)v;
        else if (sec == 1) Kout[idx] = v;
        else               Vout[idx] = v;
      }
    }
  }
}

// ---------------------------------------------------------------------------
// Kernel 2: causal flash attention. One wave per (b,h, 16-query tile).
// Key tiles of 32; S(16x32) = two C-tiles; P re-laid via per-wave LDS to an
// A-fragment; O(16x128) = 8 C-tiles accumulated with WMMA.
// V tile (32x128 f32, 16KB/wave) is staged Global->LDS with the gfx1250 async
// copy engine at the top of the iteration and consumed after s_wait_asynccnt,
// hiding its latency behind the 8 QK^T WMMAs + softmax. No block barriers.
// Grid (B*H=32, 128/2=64), block 64 (2 waves).
// ---------------------------------------------------------------------------
__global__ __launch_bounds__(64) void attn_kernel(
    const __bf16* __restrict__ Qbf, const float* __restrict__ Kf,
    const float* __restrict__ Vf, __bf16* __restrict__ attbf) {
  __shared__ __align__(16) float sP[2 * 16 * 32];     // per-wave 16x32 P staging
#if HAVE_ASYNC_LDS
  __shared__ __align__(16) float sV[2 * 32 * kHD];    // per-wave 32x128 V tile
#endif
  int lane = threadIdx.x & 31, w = threadIdx.x >> 5;
  int bh = blockIdx.x;                 // 0..31
  int qt = blockIdx.y * 2 + w;         // 0..127
  int bidx = bh >> 4, h = bh & 15;
  int q0 = qt * 16;
  int n = lane & 15, half = lane >> 4;
  const float scale = 0.08838834764831845f;  // 1/sqrt(128)

  const __bf16* Qh = Qbf + (size_t)bh * kT * kHD;
  const float*  Kh = Kf  + (size_t)bh * kT * kHD;
  const float*  Vh = Vf  + (size_t)bh * kT * kHD;
  float* myP = &sP[w * 16 * 32];
#if HAVE_ASYNC_LDS
  float* myV = &sV[w * 32 * kHD];
#endif

  v16bf qf[4];
#pragma unroll
  for (int c = 0; c < 4; ++c) qf[c] = load_a_frag_bf16(Qh, q0, kHD, c * 32, lane);

  v8f o[8] = {};                 // O: 16x128 f32 as 8 C-tiles
  float m_i[8], l_i[8];
#pragma unroll
  for (int r = 0; r < 8; ++r) { m_i[r] = -1e30f; l_i[r] = 0.0f; }

  for (int kt = 0; kt * 32 <= q0 + 15; ++kt) {
    int k0 = kt * 32;

#if HAVE_ASYNC_LDS
    // Kick off async Global->LDS copy of the V tile (32 rows x 512B);
    // each op moves 32 lanes x 16B. Overlaps with QK^T + softmax below.
#pragma unroll
    for (int row = 0; row < 32; ++row) {
      __builtin_amdgcn_global_load_async_to_lds_b128(
          (gv4i*)(Vh + (size_t)(k0 + row) * kHD + lane * 4),
          (lv4i*)(myV + row * kHD + lane * 4), 0, 0);
    }
#endif
    // prefetch next key tile's K rows (speculative; dropped if OOB)
    __builtin_prefetch(Kh + (size_t)(k0 + 32 + n + half * 16) * kHD, 0, 1);

    // S = Q K^T : two 16-col key tiles
    v8f s0 = {}, s1 = {};
#pragma unroll
    for (int c = 0; c < 4; ++c) {
      v16bf kb0 = load_a_frag_f32(Kh, k0,      kHD, c * 32, lane);
      v16bf kb1 = load_a_frag_f32(Kh, k0 + 16, kHD, c * 32, lane);
      s0 = wmma_bf16(qf[c], kb0, s0);
      s1 = wmma_bf16(qf[c], kb1, s1);
    }
    // scale + causal mask (C layout: elem r -> row r+8*half, lane -> col n)
#pragma unroll
    for (int r = 0; r < 8; ++r) {
      int query = q0 + r + half * 8;
      float a0 = s0[r] * scale, a1 = s1[r] * scale;
      if (k0 + n > query)      a0 = -1e30f;
      if (k0 + 16 + n > query) a1 = -1e30f;
      s0[r] = a0; s1[r] = a1;
    }
    // online softmax; reductions stay within each 16-lane half (xor<16)
#pragma unroll
    for (int r = 0; r < 8; ++r) {
      float mx = fmaxf(s0[r], s1[r]);
      mx = fmaxf(mx, __shfl_xor(mx, 1));
      mx = fmaxf(mx, __shfl_xor(mx, 2));
      mx = fmaxf(mx, __shfl_xor(mx, 4));
      mx = fmaxf(mx, __shfl_xor(mx, 8));
      float mnew  = fmaxf(m_i[r], mx);
      float alpha = __expf(m_i[r] - mnew);
      float p0 = __expf(s0[r] - mnew);
      float p1 = __expf(s1[r] - mnew);
      float rs = p0 + p1;
      rs += __shfl_xor(rs, 1);
      rs += __shfl_xor(rs, 2);
      rs += __shfl_xor(rs, 4);
      rs += __shfl_xor(rs, 8);
      l_i[r] = l_i[r] * alpha + rs;
      m_i[r] = mnew;
#pragma unroll
      for (int tn = 0; tn < 8; ++tn) o[tn][r] *= alpha;
      int m = r + half * 8;
      myP[m * 32 + n]      = p0;   // C-layout -> [row][key] in LDS
      myP[m * 32 + 16 + n] = p1;
    }
    asm volatile("s_wait_dscnt 0" ::: "memory");   // intra-wave LDS RAW (P)
    // P back as A-fragment (16x32), then O += P @ V
    v16bf pa = load_a_frag_f32(myP, 0, 32, 0, lane);
#if HAVE_ASYNC_LDS
    asm volatile("s_wait_asynccnt 0" ::: "memory");  // V tile landed in LDS
#pragma unroll
    for (int tn = 0; tn < 8; ++tn) {
      v16bf vb = load_b_frag_f32(myV, 0, kHD, tn * 16, lane);   // ds reads
      o[tn] = wmma_bf16(pa, vb, o[tn]);
    }
#else
#pragma unroll
    for (int tn = 0; tn < 8; ++tn) {
      v16bf vb = load_b_frag_f32(Vh, k0, kHD, tn * 16, lane);
      o[tn] = wmma_bf16(pa, vb, o[tn]);
    }
#endif
  }

  // normalize and write att output as [B, T, H*D] bf16 (row-major for proj GEMM)
#pragma unroll
  for (int r = 0; r < 8; ++r) {
    float inv = 1.0f / l_i[r];
    int t = q0 + r + half * 8;
    __bf16* dst = attbf + ((size_t)bidx * kT + t) * kDIM + h * kHD + n;
#pragma unroll
    for (int tn = 0; tn < 8; ++tn) dst[tn * 16] = (__bf16)(o[tn][r] * inv);
  }
}

// ---------------------------------------------------------------------------
// Kernel 3: out = att @ W_out + b_out. 32x64 per wave.
// Grid (2048/64/4 = 8, 4096/32 = 128), block 128.
// ---------------------------------------------------------------------------
__global__ __launch_bounds__(128) void proj_kernel(
    const __bf16* __restrict__ att, const float* __restrict__ Wout,
    const float* __restrict__ bout, float* __restrict__ out) {
  int lane = threadIdx.x & 31, w = threadIdx.x >> 5;
  int col0 = (blockIdx.x * 4 + w) * 64;   // 0..1984
  int row0 = blockIdx.y * 32;             // 0..4064
  int n = lane & 15, half = lane >> 4;

  v8f acc[2][4] = {};
  for (int kb = 0; kb < kDIM; kb += 32) {
    v16bf a0 = load_a_frag_bf16(att, row0,      kDIM, kb, lane);
    v16bf a1 = load_a_frag_bf16(att, row0 + 16, kDIM, kb, lane);
    __builtin_prefetch(att + (size_t)(row0 + n) * kDIM + kb + 32, 0, 1);
    __builtin_prefetch(Wout + (size_t)(kb + 32 + half * 8) * kDIM + col0 + n * 4, 0, 1);
#pragma unroll
    for (int j = 0; j < 4; ++j) {
      v16bf b = load_b_frag_f32(Wout, kb, kDIM, col0 + j * 16, lane);
      acc[0][j] = wmma_bf16(a0, b, acc[0][j]);
      acc[1][j] = wmma_bf16(a1, b, acc[1][j]);
    }
  }

#pragma unroll
  for (int j = 0; j < 4; ++j) {
    int col = col0 + j * 16 + n;
    float bias = bout[col];
#pragma unroll
    for (int rb = 0; rb < 2; ++rb) {
#pragma unroll
      for (int r = 0; r < 8; ++r) {
        int row = row0 + rb * 16 + r + half * 8;
        out[(size_t)row * kDIM + col] = acc[rb][j][r] + bias;
      }
    }
  }
}

// ---------------------------------------------------------------------------
extern "C" void kernel_launch(void* const* d_in, const int* in_sizes, int n_in,
                              void* d_out, int out_size, void* d_ws, size_t ws_size,
                              hipStream_t stream) {
  (void)in_sizes; (void)n_in; (void)out_size; (void)ws_size;
  const float* x    = (const float*)d_in[0];
  // d_in[1] is the causal mask (bool) — causality is applied analytically.
  const float* Wqkv = (const float*)d_in[2];
  const float* bqkv = (const float*)d_in[3];
  const float* Wout = (const float*)d_in[4];
  const float* bout = (const float*)d_in[5];

  float* out  = (float*)d_out;                       // [B,T,C]
  float* Kout = out  + (size_t)kBT * kDIM;           // [B,H,T,D]
  float* Vout = Kout + (size_t)kBT * kDIM;           // [B,H,T,D]

  __bf16* Qbf   = (__bf16*)d_ws;                     // [B,H,T,D] bf16 (16 MB)
  __bf16* attbf = Qbf + (size_t)kBT * kDIM;          // [B,T,C]   bf16 (16 MB)

  qkv_kernel <<<dim3(24, 128), 128, 0, stream>>>(x, Wqkv, bqkv, Qbf, Kout, Vout);
  attn_kernel<<<dim3(32, 64),  64,  0, stream>>>(Qbf, Kout, Vout, attbf);
  proj_kernel<<<dim3(8, 128),  128, 0, stream>>>(attbf, Wout, bout, out);
}
